// myModel_42296837931317
// MI455X (gfx1250) — matrix operand order repaired and checked by reading, not compiled, not run on hardware
//
#include <hip/hip_runtime.h>
#include <hip/hip_bf16.h>
#include <math.h>

// Problem constants: B=4, LX=512, K=32, LG=65, H=768, NH=8, HD=96, DFF=2048, NCLS=18
#define B_   4
#define LX_  512
#define KEV  32
#define LG_  65
#define H_   768
#define NH_  8
#define HD_  96
#define DFF_ 2048
#define NCLS_ 18

typedef float  v8f  __attribute__((ext_vector_type(8)));
typedef __bf16 v16bf __attribute__((ext_vector_type(16)));
typedef int    v4i  __attribute__((ext_vector_type(4)));

union ABfrag { unsigned int u[8]; v16bf v; };

__device__ __forceinline__ unsigned short f2bf(float f) {
    unsigned int u = __float_as_uint(f);
    u += 0x7fffu + ((u >> 16) & 1u);            // round-to-nearest-even
    return (unsigned short)(u >> 16);
}
__device__ __forceinline__ unsigned int packbf(float lo, float hi) {
    return (unsigned int)f2bf(lo) | ((unsigned int)f2bf(hi) << 16);
}
// A-matrix 16x32 bf16 pair->K mapping (ISA 7.12.2): VGPR pair p, lane-half h
__device__ __forceinline__ int pk_a(int p, int half) { return ((p >> 2) << 4) + ((p & 3) << 1) + (half << 3); }
// B-matrix 32x16 bf16 pair->K mapping (same as A, ISA 7.12.2)
__device__ __forceinline__ int pk_b(int p, int half) { return ((p >> 2) << 4) + ((p & 3) << 1) + (half << 3); }
__device__ __forceinline__ float sigmoidf_(float x) { return 1.0f / (1.0f + __expf(-x)); }

// ---- permlane16-based xor swap within each 16-lane row (wave32: two rows) ----
constexpr unsigned selw(int base, int mask) {
    unsigned w = 0;
    for (int i = 0; i < 8; ++i) w |= ((unsigned)(((base + i) ^ mask) & 15)) << (4 * i);
    return w;
}
template<int MASK>
__device__ __forceinline__ float xswap(float v) {
#if __has_builtin(__builtin_amdgcn_permlane16)
    constexpr unsigned s0 = selw(0, MASK), s1 = selw(8, MASK);
    unsigned r = __builtin_amdgcn_permlane16(__float_as_uint(v), __float_as_uint(v), s0, s1, false, false);
    return __uint_as_float(r);
#else
    return __shfl_xor(v, MASK, 32);
#endif
}

// ---- async global->LDS (gfx1250 ASYNCcnt path), guarded with fallback ----
#if __has_builtin(__builtin_amdgcn_global_load_async_to_lds_b128)
#define HAVE_ASYNC_LDS 1
#else
#define HAVE_ASYNC_LDS 0
#endif

__device__ __forceinline__ void wait_async0() {
#if __has_builtin(__builtin_amdgcn_s_wait_asynccnt)
    __builtin_amdgcn_s_wait_asynccnt(0);
#elif HAVE_ASYNC_LDS
    asm volatile("s_wait_asynccnt 0" ::: "memory");
#endif
}

__device__ __forceinline__ void async_b128(const void* gsrc, void* ldst) {
#if HAVE_ASYNC_LDS
    __builtin_amdgcn_global_load_async_to_lds_b128(
        (__attribute__((address_space(1))) v4i*)(unsigned long long)gsrc,
        (__attribute__((address_space(3))) v4i*)ldst, 0, 0);
#else
    const unsigned int* g32 = (const unsigned int*)gsrc;
    unsigned int* l32 = (unsigned int*)ldst;
#pragma unroll
    for (int q = 0; q < 4; ++q) l32[q] = g32[q];
#endif
}

// ---------------------------------------------------------------- RoPE (interleaved)
__global__ __launch_bounds__(384)
void rope_k(const float* __restrict__ x, float* __restrict__ y) {
    int r = blockIdx.x;                 // b*512 + l
    int i = threadIdx.x;                // pair index 0..383
    int l = r & (LX_ - 1);
    float inv = __expf(-9.210340371976184f * (2.0f * (float)i) / 768.0f); // 10000^{-2i/768}
    float ang = (float)l * inv;
    float c = __cosf(ang), sn = __sinf(ang);
    size_t base = (size_t)r * H_ + 2 * i;
    float x0 = x[base], x1 = x[base + 1];
    y[base]     = x0 * c - x1 * sn;
    y[base + 1] = x1 * c + x0 * sn;
}

// ---------------------------------------------------------------- weight convert+transpose
// WT[n*K + k] = bf16(W[k*N + n]); 32x32 LDS tile transpose, one-time cost (~19MB -> L2)
__global__ __launch_bounds__(256)
void convT_bf16(const float* __restrict__ W, unsigned short* __restrict__ WT, int K, int N) {
    __shared__ float t[32][33];
    int n0 = blockIdx.x * 32, k0 = blockIdx.y * 32;
    int tx = threadIdx.x & 31, ty = threadIdx.x >> 5;
#pragma unroll
    for (int i = 0; i < 4; ++i)
        t[ty + i * 8][tx] = W[(size_t)(k0 + ty + i * 8) * N + n0 + tx];
    __syncthreads();
#pragma unroll
    for (int i = 0; i < 4; ++i)
        WT[(size_t)(n0 + ty + i * 8) * K + k0 + tx] = f2bf(t[tx][ty + i * 8]);
}

// ---------------------------------------------------------------- Generic WMMA GEMM
// C[M,N] = epi( [A1|A2][M,K] @ W[K,N] + bias ), W given as bf16 W^T [N][K] (row stride ldw).
// EPI: 0=none, 1=relu, 2=sigmoid-gate blend of GX/GY.
// 256 threads (8 wave32 waves), tile 128x64, BK=32, wave computes 32x32 (2x2 WMMA).
// B tile staged via one global_load_async_to_lds_b128 per lane (16B chunks, aligned stride 40).
template<int EPI>
__global__ __launch_bounds__(256)
void gemm_bf16_wmma(const float* __restrict__ A1, const float* __restrict__ A2, int K1,
                    const unsigned short* __restrict__ WT, int ldw,
                    const float* __restrict__ bias,
                    float* __restrict__ C, int M, int N, int K,
                    const float* __restrict__ GX, const float* __restrict__ GY) {
    __shared__ unsigned short As[128 * 34];   // [row][k], padded stride
    __shared__ unsigned short Bs[64 * 40];    // [col][k], stride 40 (80B, 16B-aligned chunks)
    const int tid = threadIdx.x;
    const int lane = tid & 31, wid = tid >> 5;
    const int half = lane >> 4, rm = lane & 15;
    const int wm = wid & 3, wn = wid >> 2;    // 4x2 wave grid
    const int brow = blockIdx.y * 128, bcol = blockIdx.x * 64;
    const int K2 = K - K1;
    const int bcol2 = tid >> 2, bchunk = tid & 3;   // B-stage: 64 cols x 4 x 16B

    v8f acc[2][2] = {};
    for (int k0 = 0; k0 < K; k0 += 32) {
        __syncthreads();
        // --- B tile: async copy bf16 W^T (ASYNCcnt path) ---
        async_b128(WT + (size_t)(bcol + bcol2) * ldw + k0 + bchunk * 8,
                   &Bs[bcol2 * 40 + bchunk * 8]);
        // --- A tile 128x32: f32 -> bf16 (activations stay f32 in memory) ---
#pragma unroll
        for (int i = 0; i < 16; ++i) {
            int idx = tid + i * 256;
            int r = idx >> 5, kk = idx & 31;
            int gr = brow + r, gk = k0 + kk;
            float v = 0.0f;
            if (gr < M) v = (gk < K1) ? A1[(size_t)gr * K1 + gk]
                                      : A2[(size_t)gr * K2 + (gk - K1)];
            As[r * 34 + kk] = f2bf(v);
        }
        wait_async0();
        __syncthreads();
        ABfrag a[2], b[2];
#pragma unroll
        for (int t = 0; t < 2; ++t) {
            int row = wm * 32 + t * 16 + rm;
            int col = wn * 32 + t * 16 + rm;
#pragma unroll
            for (int p = 0; p < 8; ++p) {
                a[t].u[p] = *(const unsigned int*)&As[row * 34 + pk_a(p, half)];
                b[t].u[p] = *(const unsigned int*)&Bs[col * 40 + pk_b(p, half)];
            }
        }
#pragma unroll
        for (int i = 0; i < 2; ++i)
#pragma unroll
            for (int j = 0; j < 2; ++j)
                acc[i][j] = __builtin_amdgcn_wmma_f32_16x16x32_bf16(
                    false, a[i].v, false, b[j].v, (short)0, acc[i][j], false, false);
    }
    // epilogue: C layout row = r + 8*half, col = lane&15
#pragma unroll
    for (int i = 0; i < 2; ++i) {
#pragma unroll
        for (int j = 0; j < 2; ++j) {
            int col = bcol + wn * 32 + j * 16 + rm;
            float bv = bias ? bias[col] : 0.0f;
#pragma unroll
            for (int r = 0; r < 8; ++r) {
                int row = brow + wm * 32 + i * 16 + half * 8 + r;
                if (row < M) {
                    float v = acc[i][j][r] + bv;
                    if (EPI == 1) v = fmaxf(v, 0.0f);
                    if (EPI == 2) {
                        float gt = sigmoidf_(v);
                        float xv = GX[(size_t)row * N + col];
                        float yv = GY[(size_t)row * N + col];
                        v = gt * xv + (1.0f - gt) * yv;
                    }
                    C[(size_t)row * N + col] = v;
                }
            }
        }
    }
}

// ---------------------------------------------------------------- Flash-style MHA (WMMA)
// Block: (b, h, 64-row q tile); 128 threads = 4 waves, 16 q rows per wave.
// Streams 32-key blocks: S = Q K^T (6 WMMA), online softmax (permlane16), O += P V (6 WMMA).
__global__ __launch_bounds__(128)
void mha_wmma(const float* __restrict__ qkv, float* __restrict__ ao) {
    __shared__ unsigned short Ks[32 * 98];        // [key][d], d in 0..95
    __shared__ unsigned short Vs[96 * 34];        // [d][key] transposed
    __shared__ unsigned short Ps[4 * 16 * 34];    // per-wave P tile 16x32
    const int tid = threadIdx.x;
    const int lane = tid & 31, w = tid >> 5;
    const int half = lane >> 4, rm = lane & 15;
    const int qt = blockIdx.x & 7;                // 8 q tiles of 64 rows
    const int bh = blockIdx.x >> 3;
    const int b = bh >> 3, h = bh & 7;
    const float scale = 0.10206207261596575f;     // 1/sqrt(96), folded into Q

    ABfrag qa[3];                                  // Q fragments, K-dim = 96 (3 steps)
    {
        int row = qt * 64 + w * 16 + rm;
        const float* qp = qkv + ((size_t)(b * LX_ + row)) * (3 * H_) + h * HD_;
#pragma unroll
        for (int ks = 0; ks < 3; ++ks)
#pragma unroll
            for (int p = 0; p < 8; ++p) {
                int d = ks * 32 + pk_a(p, half);
                qa[ks].u[p] = packbf(qp[d] * scale, qp[d + 1] * scale);
            }
    }

    v8f of[6] = {};
    float mi[8], li[8];
#pragma unroll
    for (int r = 0; r < 8; ++r) { mi[r] = -1e30f; li[r] = 0.0f; }

    for (int j = 0; j < LX_; j += 32) {
        __syncthreads();
        for (int i = tid; i < 32 * 96; i += 128) {    // stage K,V block (coalesced f32->bf16)
            int key = i / 96, d = i - key * 96;
            const float* kp = qkv + ((size_t)(b * LX_ + j + key)) * (3 * H_) + h * HD_;
            Ks[key * 98 + d] = f2bf(kp[H_ + d]);      // K at +768
            Vs[d * 34 + key] = f2bf(kp[2 * H_ + d]);  // V at +1536, transposed store
        }
        __syncthreads();
        v8f sc[2] = {};
#pragma unroll
        for (int nt = 0; nt < 2; ++nt)
#pragma unroll
            for (int ks = 0; ks < 3; ++ks) {
                ABfrag kb;                            // B frag of K^T: kdim=d, col=key
#pragma unroll
                for (int p = 0; p < 8; ++p) {
                    int d = ks * 32 + pk_b(p, half);
                    kb.u[p] = *(const unsigned int*)&Ks[(nt * 16 + rm) * 98 + d];
                }
                sc[nt] = __builtin_amdgcn_wmma_f32_16x16x32_bf16(
                    false, qa[ks].v, false, kb.v, (short)0, sc[nt], false, false);
            }
        // online softmax; row r lives in a 16-lane row -> permlane16 xor reductions
        float al[8];
#pragma unroll
        for (int r = 0; r < 8; ++r) {
            float s0 = sc[0][r], s1 = sc[1][r];
            float t = fmaxf(s0, s1);
            t = fmaxf(t, xswap<1>(t));
            t = fmaxf(t, xswap<2>(t));
            t = fmaxf(t, xswap<4>(t));
            t = fmaxf(t, xswap<8>(t));
            float mnew = fmaxf(mi[r], t);
            float a = __expf(mi[r] - mnew);
            float p0 = __expf(s0 - mnew), p1 = __expf(s1 - mnew);
            float rs = p0 + p1;
            rs += xswap<1>(rs);
            rs += xswap<2>(rs);
            rs += xswap<4>(rs);
            rs += xswap<8>(rs);
            li[r] = li[r] * a + rs;
            mi[r] = mnew; al[r] = a;
            // C-layout -> LDS (per-wave private region; DS ops in-order per wave)
            Ps[w * 544 + (half * 8 + r) * 34 + rm]      = f2bf(p0);
            Ps[w * 544 + (half * 8 + r) * 34 + 16 + rm] = f2bf(p1);
        }
#pragma unroll
        for (int nt = 0; nt < 6; ++nt)
#pragma unroll
            for (int r = 0; r < 8; ++r) of[nt][r] *= al[r];
        ABfrag pa;                                    // reload P in A-fragment layout
#pragma unroll
        for (int p = 0; p < 8; ++p)
            pa.u[p] = *(const unsigned int*)&Ps[w * 544 + rm * 34 + pk_a(p, half)];
#pragma unroll
        for (int nt = 0; nt < 6; ++nt) {
            ABfrag vb;                                // B frag of V: kdim=key, col=d
#pragma unroll
            for (int p = 0; p < 8; ++p)
                vb.u[p] = *(const unsigned int*)&Vs[(nt * 16 + rm) * 34 + pk_b(p, half)];
            of[nt] = __builtin_amdgcn_wmma_f32_16x16x32_bf16(
                false, pa.v, false, vb.v, (short)0, of[nt], false, false);
        }
    }
#pragma unroll
    for (int nt = 0; nt < 6; ++nt)
#pragma unroll
        for (int r = 0; r < 8; ++r) {
            int row = qt * 64 + w * 16 + half * 8 + r;
            ao[((size_t)(b * LX_ + row)) * H_ + h * HD_ + nt * 16 + rm] = of[nt][r] / li[r];
        }
}

// ---------------------------------------------------------------- residual + LayerNorm
__global__ __launch_bounds__(256)
void add_ln(const float* __restrict__ A, const float* __restrict__ Bv,
            const float* __restrict__ g, const float* __restrict__ be,
            float* __restrict__ O) {
    __shared__ float red[256];
    int row = blockIdx.x, tid = threadIdx.x;
    float v[3]; float sum = 0.0f;
#pragma unroll
    for (int i = 0; i < 3; ++i) {
        int d = tid + i * 256;
        v[i] = A[(size_t)row * H_ + d] + Bv[(size_t)row * H_ + d];
        sum += v[i];
    }
    red[tid] = sum; __syncthreads();
    for (int st = 128; st > 0; st >>= 1) { if (tid < st) red[tid] += red[tid + st]; __syncthreads(); }
    float mean = red[0] * (1.0f / 768.0f);
    __syncthreads();
    float sq = 0.0f;
#pragma unroll
    for (int i = 0; i < 3; ++i) { float dd = v[i] - mean; sq += dd * dd; }
    red[tid] = sq; __syncthreads();
    for (int st = 128; st > 0; st >>= 1) { if (tid < st) red[tid] += red[tid + st]; __syncthreads(); }
    float rstd = rsqrtf(red[0] * (1.0f / 768.0f) + 1e-5f);
#pragma unroll
    for (int i = 0; i < 3; ++i) {
        int d = tid + i * 256;
        O[(size_t)row * H_ + d] = (v[i] - mean) * rstd * g[d] + be[d];
    }
}

// ---------------------------------------------------------------- classification head (N=18, tiny)
__global__ __launch_bounds__(256)
void cls_head(const float* __restrict__ y2, const float* __restrict__ wcls,
              const float* __restrict__ bcls, float* __restrict__ argout) {
    __shared__ float rowbuf[H_];
    int r = blockIdx.x, tid = threadIdx.x;
    for (int d = tid; d < H_; d += 256) rowbuf[d] = y2[(size_t)r * H_ + d];
    __syncthreads();
    if (tid < NCLS_) {
        float acc = bcls[tid];
        for (int d = 0; d < H_; ++d) acc += rowbuf[d] * wcls[d * NCLS_ + tid];
        int b = r >> 9, l = r & (LX_ - 1);
        argout[((size_t)(b * NCLS_ + tid)) * LX_ + l] = sigmoidf_(acc);
    }
}

// ---------------------------------------------------------------- fusion cross-attention (LG=65)
__global__ __launch_bounds__(128)
void fusion_attn(const float* __restrict__ qf, const float* __restrict__ kv,
                 float* __restrict__ ga) {
    __shared__ float qrow[H_];
    __shared__ float pm[LG_];
    int r = blockIdx.x, tid = threadIdx.x;
    int b = r >> 9;
    for (int d = tid; d < H_; d += 128) qrow[d] = qf[(size_t)r * H_ + d];
    __syncthreads();
    const float rs = 0.036084391824351615f;   // 1/sqrt(768)
    if (tid < LG_) {
        const float* kf = kv + ((size_t)(b * LG_ + tid)) * (2 * H_);
        float a = 0.0f;
        for (int d = 0; d < H_; ++d) a += qrow[d] * kf[d];
        pm[tid] = a * rs;
    }
    __syncthreads();
    float mx = -1e30f;
    for (int m = 0; m < LG_; ++m) mx = fmaxf(mx, pm[m]);
    float sum = 0.0f;
    for (int m = 0; m < LG_; ++m) sum += __expf(pm[m] - mx);
    __syncthreads();
    if (tid < LG_) pm[tid] = __expf(pm[tid] - mx) / sum;
    __syncthreads();
    for (int d = tid; d < H_; d += 128) {
        float a = 0.0f;
        for (int m = 0; m < LG_; ++m)
            a += pm[m] * kv[((size_t)(b * LG_ + m)) * (2 * H_) + H_ + d];
        ga[(size_t)r * H_ + d] = a;
    }
}

// ---------------------------------------------------------------- final broadcast blend (BW-bound)
// h[b,l,k,:] = sig(A2[b,l,:] + Cs[b,k,:]) * h1[b,l,:] + (1-sig)*s[b,k,:]
__global__ __launch_bounds__(192)
void final_fuse(const float* __restrict__ a2, const float* __restrict__ cs,
                const float* __restrict__ h1, const float* __restrict__ s,
                float* __restrict__ hout) {
    int idx = blockIdx.x;                 // (b*LX + l)*KEV + k
    int k = idx & (KEV - 1), bl = idx >> 5;
    int b = bl >> 9;
    int t = threadIdx.x;                  // 192 threads x float4 = 768
    const float4 av = ((const float4*)(a2 + (size_t)bl * H_))[t];
    const float4 cv = ((const float4*)(cs + ((size_t)(b * KEV + k)) * H_))[t];
    const float4 hv = ((const float4*)(h1 + (size_t)bl * H_))[t];
    const float4 sv = ((const float4*)(s  + ((size_t)(b * KEV + k)) * H_))[t];
    float4 o; float g;
    g = sigmoidf_(av.x + cv.x); o.x = g * hv.x + (1.0f - g) * sv.x;
    g = sigmoidf_(av.y + cv.y); o.y = g * hv.y + (1.0f - g) * sv.y;
    g = sigmoidf_(av.z + cv.z); o.z = g * hv.z + (1.0f - g) * sv.z;
    g = sigmoidf_(av.w + cv.w); o.w = g * hv.w + (1.0f - g) * sv.w;
    ((float4*)(hout + (size_t)idx * H_))[t] = o;
}

extern "C" void kernel_launch(void* const* d_in, const int* in_sizes, int n_in,
                              void* d_out, int out_size, void* d_ws, size_t ws_size,
                              hipStream_t stream) {
    (void)in_sizes; (void)n_in; (void)out_size; (void)ws_size;
    const float* x        = (const float*)d_in[0];
    const float* s        = (const float*)d_in[1];
    const float* g        = (const float*)d_in[2];
    const float* enc_wqkv = (const float*)d_in[3];
    const float* enc_bqkv = (const float*)d_in[4];
    const float* enc_wo   = (const float*)d_in[5];
    const float* enc_bo   = (const float*)d_in[6];
    const float* ln1_g    = (const float*)d_in[7];
    const float* ln1_b    = (const float*)d_in[8];
    const float* ff_w1    = (const float*)d_in[9];
    const float* ff_b1    = (const float*)d_in[10];
    const float* ff_w2    = (const float*)d_in[11];
    const float* ff_b2    = (const float*)d_in[12];
    const float* ln2_g    = (const float*)d_in[13];
    const float* ln2_b    = (const float*)d_in[14];
    const float* cls_w    = (const float*)d_in[15];
    const float* cls_b    = (const float*)d_in[16];
    const float* af_wq    = (const float*)d_in[17];
    const float* af_bq    = (const float*)d_in[18];
    const float* af_wkv   = (const float*)d_in[19];
    const float* af_bkv   = (const float*)d_in[20];
    const float* g1_w     = (const float*)d_in[21];
    const float* g1_b     = (const float*)d_in[22];
    const float* g2_w     = (const float*)d_in[23];
    const float* g2_b     = (const float*)d_in[24];

    float* ws = (float*)d_ws;
    const size_t R = (size_t)B_ * LX_;      // 2048 rows
    size_t o = 0;
    float* y   = ws + o; o += R * H_;
    float* qkv = ws + o; o += R * 3 * H_;
    float* at  = ws + o; o += R * H_;
    float* ao  = ws + o; o += R * H_;
    float* y1  = ws + o; o += R * H_;
    float* f1  = ws + o; o += R * DFF_;
    float* f2  = ws + o; o += R * H_;
    float* y2  = ws + o; o += R * H_;
    float* qf  = ws + o; o += R * H_;
    float* kvb = ws + o; o += (size_t)B_ * LG_ * 2 * H_;
    float* gab = ws + o; o += R * H_;
    float* h1  = ws + o; o += R * H_;
    float* a2  = ws + o; o += R * H_;
    float* cs  = ws + o; o += (size_t)B_ * KEV * H_;

    // bf16 transposed weight shadows (one-time convert, lives in L2)
    unsigned short* wsh = (unsigned short*)(ws + o);
    size_t oh = 0;
    unsigned short* wt_qkv = wsh + oh; oh += (size_t)H_ * 3 * H_;     // [2304][768]
    unsigned short* wt_wo  = wsh + oh; oh += (size_t)H_ * H_;         // [768][768]
    unsigned short* wt_f1  = wsh + oh; oh += (size_t)H_ * DFF_;       // [2048][768]
    unsigned short* wt_f2  = wsh + oh; oh += (size_t)DFF_ * H_;       // [768][2048]
    unsigned short* wt_aq  = wsh + oh; oh += (size_t)H_ * H_;         // [768][768]
    unsigned short* wt_akv = wsh + oh; oh += (size_t)H_ * 2 * H_;     // [1536][768]
    unsigned short* wt_g1  = wsh + oh; oh += (size_t)2 * H_ * H_;     // [768][1536]
    unsigned short* wt_g2  = wsh + oh; oh += (size_t)2 * H_ * H_;     // [768][1536]

    float* arg_out = (float*)d_out;
    float* hout    = arg_out + (size_t)B_ * NCLS_ * LX_;

    // ---- one-time weight convert+transpose (f32 -> bf16 [N][K]) ----
    convT_bf16<<<dim3(3 * H_ / 32, H_ / 32), 256, 0, stream>>>(enc_wqkv, wt_qkv, H_, 3 * H_);
    convT_bf16<<<dim3(H_ / 32, H_ / 32), 256, 0, stream>>>(enc_wo, wt_wo, H_, H_);
    convT_bf16<<<dim3(DFF_ / 32, H_ / 32), 256, 0, stream>>>(ff_w1, wt_f1, H_, DFF_);
    convT_bf16<<<dim3(H_ / 32, DFF_ / 32), 256, 0, stream>>>(ff_w2, wt_f2, DFF_, H_);
    convT_bf16<<<dim3(H_ / 32, H_ / 32), 256, 0, stream>>>(af_wq, wt_aq, H_, H_);
    convT_bf16<<<dim3(2 * H_ / 32, H_ / 32), 256, 0, stream>>>(af_wkv, wt_akv, H_, 2 * H_);
    convT_bf16<<<dim3(H_ / 32, 2 * H_ / 32), 256, 0, stream>>>(g1_w, wt_g1, 2 * H_, H_);
    convT_bf16<<<dim3(H_ / 32, 2 * H_ / 32), 256, 0, stream>>>(g2_w, wt_g2, 2 * H_, H_);

    // ---- argReLU encoder path ----
    rope_k<<<R, 384, 0, stream>>>(x, y);
    gemm_bf16_wmma<0><<<dim3(3 * H_ / 64, 16), 256, 0, stream>>>(
        y, nullptr, H_, wt_qkv, H_, enc_bqkv, qkv, (int)R, 3 * H_, H_, nullptr, nullptr);
    mha_wmma<<<B_ * NH_ * (LX_ / 64), 128, 0, stream>>>(qkv, at);
    gemm_bf16_wmma<0><<<dim3(H_ / 64, 16), 256, 0, stream>>>(
        at, nullptr, H_, wt_wo, H_, enc_bo, ao, (int)R, H_, H_, nullptr, nullptr);
    add_ln<<<R, 256, 0, stream>>>(y, ao, ln1_g, ln1_b, y1);
    gemm_bf16_wmma<1><<<dim3(DFF_ / 64, 16), 256, 0, stream>>>(
        y1, nullptr, H_, wt_f1, H_, ff_b1, f1, (int)R, DFF_, H_, nullptr, nullptr);
    gemm_bf16_wmma<0><<<dim3(H_ / 64, 16), 256, 0, stream>>>(
        f1, nullptr, DFF_, wt_f2, DFF_, ff_b2, f2, (int)R, H_, DFF_, nullptr, nullptr);
    add_ln<<<R, 256, 0, stream>>>(y1, f2, ln2_g, ln2_b, y2);
    cls_head<<<R, 256, 0, stream>>>(y2, cls_w, cls_b, arg_out);

    // ---- AdaptiveFusion path (gates factored over the K broadcast) ----
    gemm_bf16_wmma<0><<<dim3(H_ / 64, 16), 256, 0, stream>>>(
        x, nullptr, H_, wt_aq, H_, af_bq, qf, (int)R, H_, H_, nullptr, nullptr);
    gemm_bf16_wmma<0><<<dim3(2 * H_ / 64, (B_ * LG_ + 127) / 128), 256, 0, stream>>>(
        g, nullptr, H_, wt_akv, H_, af_bkv, kvb, B_ * LG_, 2 * H_, H_, nullptr, nullptr);
    fusion_attn<<<R, 128, 0, stream>>>(qf, kvb, gab);
    // gate1: h1 = sig([x|ga] @ g1_w + b) * x + (1-sig)*ga   (fused epilogue)
    gemm_bf16_wmma<2><<<dim3(H_ / 64, 16), 256, 0, stream>>>(
        x, gab, H_, wt_g1, 2 * H_, g1_b, h1, (int)R, H_, 2 * H_, x, gab);
    // gate2 split: A2 = h1 @ g2_w_top + g2_b ; Cs = s @ g2_w_bot
    gemm_bf16_wmma<0><<<dim3(H_ / 64, 16), 256, 0, stream>>>(
        h1, nullptr, H_, wt_g2, 2 * H_, g2_b, a2, (int)R, H_, H_, nullptr, nullptr);
    gemm_bf16_wmma<0><<<dim3(H_ / 64, 1), 256, 0, stream>>>(
        s, nullptr, H_, wt_g2 + H_, 2 * H_, nullptr, cs, B_ * KEV, H_, H_, nullptr, nullptr);
    final_fuse<<<R * KEV, 192, 0, stream>>>(a2, cs, h1, s, hout);
}
